// FIRA_3633542332736
// MI455X (gfx1250) — matrix operand order repaired and checked by reading, not compiled, not run on hardware
//
#include <hip/hip_runtime.h>
#include <math.h>

typedef _Float16 half_t;
typedef __attribute__((ext_vector_type(16))) _Float16 v16h;
typedef __attribute__((ext_vector_type(8)))  _Float16 v8h;
typedef __attribute__((ext_vector_type(8)))  float    v8f;

#define EPI_STORE 0
#define EPI_GELU  1
#define EPI_ACCUM 2

struct GemmP {
  const float* A; long long lda, sAo, sAi;
  const float* B; long long ldb, sBo, sBi;
  const float* bias;
  float*       C; long long ldc, sCo, sCi;
  const float* rowScale; int rsStride;
  int M, N, K;
  int zInner;   // z -> (z/zInner) outer, (z%zInner) inner
  int epi;
};

__device__ __forceinline__ v8h cvt8(float4 a, float4 b) {
  v8h h;
  h[0] = (half_t)a.x; h[1] = (half_t)a.y; h[2] = (half_t)a.z; h[3] = (half_t)a.w;
  h[4] = (half_t)b.x; h[5] = (half_t)b.y; h[6] = (half_t)b.z; h[7] = (half_t)b.w;
  return h;
}

// 128x64 tile, BK=32, 128 threads (4 waves). Each wave: 32 rows x 64 cols via
// 2 A-fragments x 4 B-fragments = 8 x v_wmma_f32_16x16x32_f16 per K-step.
// fp32 in global, f16 in LDS.
// REQUIRES: M%128==0, N%64==0, K%32==0, 16B-aligned A (and B when BKN==0).
// BKN==0: B is [N,K] row-major (Y = X*W^T).  BKN==1: B is [K,N] row-major (P*V).
template <int BKN>
__global__ __launch_bounds__(128) void fira_wmma_gemm(GemmP p) {
  __shared__ half_t lA[128 * 32];
  __shared__ half_t lB[64 * 32];

  const int tile_n = blockIdx.x * 64;
  const int tile_m = blockIdx.y * 128;
  const int z  = blockIdx.z;
  const int zo = z / p.zInner, zi = z % p.zInner;
  const float* __restrict__ A = p.A + zo * p.sAo + zi * p.sAi;
  const float* __restrict__ B = p.B + zo * p.sBo + zi * p.sBi;
  float*       __restrict__ C = p.C + zo * p.sCo + zi * p.sCi;

  const int tid  = threadIdx.x;
  const int lane = tid & 31;
  const int wave = tid >> 5;
  const int hs   = lane >> 4;   // half-of-wave select
  const int lr   = lane & 15;

  // cooperative-load ownership: 16 contiguous K elems per row; A has 2 rows/thread
  const int ldRow = tid >> 1;         // 0..63
  const int ldKc  = (tid & 1) * 16;   // 0 or 16

  union F8 { v8f v; float f[8]; };
  F8 acc[2][4];
  #pragma unroll
  for (int f = 0; f < 2; ++f)
    #pragma unroll
    for (int j = 0; j < 4; ++j)
      #pragma unroll
      for (int r = 0; r < 8; ++r) acc[f][j].f[r] = 0.0f;

  const float* aRow0 = A + (long long)(tile_m + ldRow)      * p.lda + ldKc;
  const float* aRow1 = A + (long long)(tile_m + ldRow + 64) * p.lda + ldKc;
  const float* bRow  = B + (long long)(tile_n + ldRow)      * p.ldb + ldKc;  // BKN==0

  for (int k0 = 0; k0 < p.K; k0 += 32) {
    // ---- A tile: two rows, each 4x float4 -> 16 halfs -> 2x b128 LDS store ----
    {
      const float4* pa0 = (const float4*)(aRow0 + k0);
      const float4* pa1 = (const float4*)(aRow1 + k0);
      float4 q0 = pa0[0], q1 = pa0[1], q2 = pa0[2], q3 = pa0[3];
      float4 r0 = pa1[0], r1 = pa1[1], r2 = pa1[2], r3 = pa1[3];
      *(v8h*)&lA[ldRow * 32 + ldKc]            = cvt8(q0, q1);
      *(v8h*)&lA[ldRow * 32 + ldKc + 8]        = cvt8(q2, q3);
      *(v8h*)&lA[(ldRow + 64) * 32 + ldKc]     = cvt8(r0, r1);
      *(v8h*)&lA[(ldRow + 64) * 32 + ldKc + 8] = cvt8(r2, r3);
    }
    // ---- B tile (stored [n][k] in LDS) ----
    if (BKN == 0) {
      const float4* pb = (const float4*)(bRow + k0);
      float4 q0 = pb[0], q1 = pb[1], q2 = pb[2], q3 = pb[3];
      *(v8h*)&lB[ldRow * 32 + ldKc]     = cvt8(q0, q1);
      *(v8h*)&lB[ldRow * 32 + ldKc + 8] = cvt8(q2, q3);
    } else {
      // transposed gather: B[k][n], unconditional strided loads
      float tmp[16];
      #pragma unroll
      for (int i = 0; i < 16; ++i)
        tmp[i] = B[(long long)(k0 + ldKc + i) * p.ldb + (tile_n + ldRow)];
      v8h h0, h1;
      #pragma unroll
      for (int i = 0; i < 8; ++i) { h0[i] = (half_t)tmp[i]; h1[i] = (half_t)tmp[8 + i]; }
      *(v8h*)&lB[ldRow * 32 + ldKc]     = h0;
      *(v8h*)&lB[ldRow * 32 + ldKc + 8] = h1;
    }
    __syncthreads();

    // A fragments (16x32 each): lane holds row; VGPR v: K pair at (v&3)*2+(v>>2)*16+hs*8
    union F16 { v16h v; unsigned u[8]; };
    F16 fa[2];
    #pragma unroll
    for (int f = 0; f < 2; ++f)
      #pragma unroll
      for (int r = 0; r < 8; ++r) {
        int kk = ((r & 3) * 2) + ((r >> 2) * 16) + hs * 8;
        fa[f].u[r] = *(const unsigned*)&lA[(wave * 32 + f * 16 + lr) * 32 + kk];
      }
    #pragma unroll
    for (int j = 0; j < 4; ++j) {
      // B fragment (32x16): lane holds col j*16+lr; VGPR v: K pair (v&3)*2+(v>>2)*16+hs*8
      F16 fb;
      #pragma unroll
      for (int r = 0; r < 8; ++r) {
        int kk = ((r & 3) * 2) + ((r >> 2) * 16) + hs * 8;
        fb.u[r] = *(const unsigned*)&lB[(j * 16 + lr) * 32 + kk];
      }
      acc[0][j].v = __builtin_amdgcn_wmma_f32_16x16x32_f16(
          false, fa[0].v, false, fb.v, (short)0, acc[0][j].v, false, false);
      acc[1][j].v = __builtin_amdgcn_wmma_f32_16x16x32_f16(
          false, fa[1].v, false, fb.v, (short)0, acc[1][j].v, false, false);
    }
    __syncthreads();
  }

  // epilogue; C layout: VGPR r -> M = r + hs*8 (within 16-row slab), N = lr
  float biasv[4];
  #pragma unroll
  for (int j = 0; j < 4; ++j)
    biasv[j] = p.bias ? p.bias[tile_n + j * 16 + lr] : 0.0f;

  #pragma unroll
  for (int f = 0; f < 2; ++f) {
    #pragma unroll
    for (int r = 0; r < 8; ++r) {
      const int m = tile_m + wave * 32 + f * 16 + hs * 8 + r;
      float* crow = C + (long long)m * p.ldc;
      float srow = 1.0f;
      if (p.epi == EPI_ACCUM && p.rowScale) srow = p.rowScale[(long long)m * p.rsStride];
      #pragma unroll
      for (int j = 0; j < 4; ++j) {
        const int n = tile_n + j * 16 + lr;
        float v = acc[f][j].f[r] + biasv[j];
        if (p.epi == EPI_GELU) {
          crow[n] = 0.5f * v * (1.0f + erff(v * 0.70710678118654752f));
        } else if (p.epi == EPI_ACCUM) {
          crow[n] += srow * v;
        } else {
          crow[n] = v;
        }
      }
    }
  }
}

__global__ void fira_embed(const int* __restrict__ ids, const float* __restrict__ tok,
                           const float* __restrict__ pos, float* __restrict__ x,
                           int T, int D, long long total) {
  long long i = (long long)blockIdx.x * blockDim.x + threadIdx.x;
  if (i >= total) return;
  long long bt = i / D;
  int d = (int)(i % D);
  int t = (int)(bt % T);
  x[i] = tok[(long long)ids[bt] * D + d] + pos[(long long)t * D + d];
}

// addMode==0: out = LN(x)*w+b ; addMode==1: out += LN(x)*w+b
__global__ __launch_bounds__(256) void fira_layernorm(const float* __restrict__ x,
                                                      const float* __restrict__ w,
                                                      const float* __restrict__ b,
                                                      float* __restrict__ out,
                                                      int D, int addMode) {
  __shared__ float red[256];
  const int row = blockIdx.x, tid = threadIdx.x;
  const float* xr = x + (long long)row * D;
  float s = 0.0f, s2 = 0.0f;
  for (int d = tid; d < D; d += 256) { float v = xr[d]; s += v; s2 += v * v; }
  red[tid] = s; __syncthreads();
  for (int o = 128; o > 0; o >>= 1) { if (tid < o) red[tid] += red[tid + o]; __syncthreads(); }
  float mean = red[0] / (float)D;
  __syncthreads();
  red[tid] = s2; __syncthreads();
  for (int o = 128; o > 0; o >>= 1) { if (tid < o) red[tid] += red[tid + o]; __syncthreads(); }
  float var = red[0] / (float)D - mean * mean;
  float inv = rsqrtf(var + 1e-5f);
  float* orow = out + (long long)row * D;
  for (int d = tid; d < D; d += 256) {
    float v = (xr[d] - mean) * inv * w[d] + b[d];
    if (addMode) orow[d] += v; else orow[d] = v;
  }
}

// in-place causal softmax on one row of scores; matches ref: s*scale, mask -> -1e9
__global__ __launch_bounds__(256) void fira_softmax_causal(float* __restrict__ scores,
                                                           int T, float scale) {
  __shared__ float red[256];
  const long long row = blockIdx.x;
  const int q = (int)(row % T);
  float* p = scores + row * (long long)T;
  const int tid = threadIdx.x;
  float mx = -3.0e38f;
  for (int k = tid; k < T; k += 256)
    if (k <= q) mx = fmaxf(mx, p[k] * scale);
  red[tid] = mx; __syncthreads();
  for (int o = 128; o > 0; o >>= 1) { if (tid < o) red[tid] = fmaxf(red[tid], red[tid + o]); __syncthreads(); }
  mx = red[0]; __syncthreads();
  float sum = 0.0f;
  for (int k = tid; k < T; k += 256)
    if (k <= q) sum += __expf(p[k] * scale - mx);
  red[tid] = sum; __syncthreads();
  for (int o = 128; o > 0; o >>= 1) { if (tid < o) red[tid] += red[tid + o]; __syncthreads(); }
  float inv = 1.0f / red[0];
  for (int k = tid; k < T; k += 256)
    p[k] = (k <= q) ? __expf(p[k] * scale - mx) * inv : 0.0f;
}

// top-2-of-4 gate with temperature; probs[m*4+e], zeros for unselected
__global__ __launch_bounds__(128) void fira_gate(const float* __restrict__ h2,
                                                 const float* __restrict__ gw,
                                                 const float* __restrict__ gb,
                                                 float* __restrict__ probs,
                                                 int D, float invTemp) {
  __shared__ float red[4][128];
  const int m = blockIdx.x, tid = threadIdx.x;
  const float* hr = h2 + (long long)m * D;
  float acc[4] = {0.0f, 0.0f, 0.0f, 0.0f};
  for (int d = tid; d < D; d += 128) {
    float hv = hr[d];
    #pragma unroll
    for (int e = 0; e < 4; ++e) acc[e] += hv * gw[e * D + d];
  }
  #pragma unroll
  for (int e = 0; e < 4; ++e) red[e][tid] = acc[e];
  __syncthreads();
  for (int o = 64; o > 0; o >>= 1) {
    if (tid < o) {
      #pragma unroll
      for (int e = 0; e < 4; ++e) red[e][tid] += red[e][tid + o];
    }
    __syncthreads();
  }
  if (tid == 0) {
    float gl[4];
    #pragma unroll
    for (int e = 0; e < 4; ++e) gl[e] = (red[e][0] + gb[e]) * invTemp;
    int i1 = 0;
    for (int e = 1; e < 4; ++e) if (gl[e] > gl[i1]) i1 = e;          // ties -> lowest idx
    int i2 = -1;
    for (int e = 0; e < 4; ++e) if (e != i1 && (i2 < 0 || gl[e] > gl[i2])) i2 = e;
    float mx = fmaxf(gl[i1], gl[i2]);
    float e1 = __expf(gl[i1] - mx), e2 = __expf(gl[i2] - mx);
    float inv = 1.0f / (e1 + e2);
    float out[4] = {0.0f, 0.0f, 0.0f, 0.0f};
    out[i1] = e1 * inv; out[i2] = e2 * inv;
    #pragma unroll
    for (int e = 0; e < 4; ++e) probs[(long long)m * 4 + e] = out[e];
  }
}

extern "C" void kernel_launch(void* const* d_in, const int* in_sizes, int n_in,
                              void* d_out, int out_size, void* d_ws, size_t ws_size,
                              hipStream_t stream) {
  const int Bx = 2, T = 1024, D = 1024, H = 16, DH = 64, L = 2, F = 4096, E = 4, V = 32000;
  const int MT = Bx * T;
  (void)in_sizes; (void)n_in; (void)out_size; (void)ws_size;

  const int*   ids    = (const int*)d_in[0];
  /* d_in[1] = causal tril mask, structurally known -> ignored */
  const float* tok    = (const float*)d_in[2];
  const float* pos    = (const float*)d_in[3];
  const float* ln1_w  = (const float*)d_in[4];
  const float* ln1_b  = (const float*)d_in[5];
  const float* wq     = (const float*)d_in[6];
  const float* bq     = (const float*)d_in[7];
  const float* wk     = (const float*)d_in[8];
  const float* bk     = (const float*)d_in[9];
  const float* wv     = (const float*)d_in[10];
  const float* bv     = (const float*)d_in[11];
  const float* wo     = (const float*)d_in[12];
  const float* bo     = (const float*)d_in[13];
  const float* ln2_w  = (const float*)d_in[14];
  const float* ln2_b  = (const float*)d_in[15];
  const float* gate_w = (const float*)d_in[16];
  const float* gate_b = (const float*)d_in[17];
  const float* e_w1   = (const float*)d_in[18];
  const float* e_b1   = (const float*)d_in[19];
  const float* e_w2   = (const float*)d_in[20];
  const float* e_b2   = (const float*)d_in[21];
  const float* pln_w  = (const float*)d_in[22];
  const float* pln_b  = (const float*)d_in[23];
  const float* lnf_w  = (const float*)d_in[24];
  const float* lnf_b  = (const float*)d_in[25];
  const float* lm_w   = (const float*)d_in[26];

  float* ws = (float*)d_ws;
  const long long ACT = (long long)MT * D;
  float* x      = ws;
  float* h      = x   + ACT;
  float* q      = h   + ACT;
  float* k      = q   + ACT;
  float* v      = k   + ACT;
  float* o      = v   + ACT;
  float* moe    = o   + ACT;
  float* h1     = moe + ACT;                    // MT * F
  float* probs  = h1  + (long long)MT * F;      // MT * E
  float* scores = probs + (long long)MT * E;    // B*H*T*T

  auto gemm = [&](const float* A, long long lda, long long sAo, long long sAi,
                  const float* Bm, long long ldb, long long sBo, long long sBi, int bKN,
                  const float* bias, float* C, long long ldc, long long sCo, long long sCi,
                  const float* rs, int rsStride,
                  int M, int N, int K, int Z, int zInner, int epi) {
    GemmP p;
    p.A = A; p.lda = lda; p.sAo = sAo; p.sAi = sAi;
    p.B = Bm; p.ldb = ldb; p.sBo = sBo; p.sBi = sBi;
    p.bias = bias;
    p.C = C; p.ldc = ldc; p.sCo = sCo; p.sCi = sCi;
    p.rowScale = rs; p.rsStride = rsStride;
    p.M = M; p.N = N; p.K = K; p.zInner = zInner; p.epi = epi;
    dim3 g((unsigned)(N / 64), (unsigned)(M / 128), (unsigned)Z);
    if (bKN == 0) fira_wmma_gemm<0><<<g, 128, 0, stream>>>(p);
    else          fira_wmma_gemm<1><<<g, 128, 0, stream>>>(p);
  };

  {
    long long total = (long long)MT * D;
    fira_embed<<<(unsigned)((total + 255) / 256), 256, 0, stream>>>(ids, tok, pos, x, T, D, total);
  }

  const float scale = 0.125f;  // 1/sqrt(DH=64)

  for (int l = 0; l < L; ++l) {
    const long long lD = (long long)l * D;
    fira_layernorm<<<MT, 256, 0, stream>>>(x, ln1_w + lD, ln1_b + lD, h, D, 0);

    gemm(h, D, 0, 0, wq + (long long)l * D * D, D, 0, 0, 0, bq + lD,
         q, D, 0, 0, nullptr, 0, MT, D, D, 1, 1, EPI_STORE);
    gemm(h, D, 0, 0, wk + (long long)l * D * D, D, 0, 0, 0, bk + lD,
         k, D, 0, 0, nullptr, 0, MT, D, D, 1, 1, EPI_STORE);
    gemm(h, D, 0, 0, wv + (long long)l * D * D, D, 0, 0, 0, bv + lD,
         v, D, 0, 0, nullptr, 0, MT, D, D, 1, 1, EPI_STORE);

    // scores[b,h] = Q[b,h] * K[b,h]^T   (A/B base = b*T*D + h*DH, ld = D)
    gemm(q, D, (long long)T * D, DH,
         k, D, (long long)T * D, DH, 0, nullptr,
         scores, T, (long long)H * T * T, (long long)T * T,
         nullptr, 0, T, T, DH, Bx * H, H, EPI_STORE);

    fira_softmax_causal<<<(unsigned)(Bx * H * T), 256, 0, stream>>>(scores, T, scale);

    // o[b,h] = P[b,h] * V[b,h]  (B is [K=T, N=DH] row-major -> bKN=1)
    gemm(scores, T, (long long)H * T * T, (long long)T * T,
         v, D, (long long)T * D, DH, 1, nullptr,
         o, D, (long long)T * D, DH,
         nullptr, 0, T, DH, T, Bx * H, H, EPI_STORE);

    // x += o * Wo^T + bo (residual via ACCUM, scale=1)
    gemm(o, D, 0, 0, wo + (long long)l * D * D, D, 0, 0, 0, bo + lD,
         x, D, 0, 0, nullptr, 0, MT, D, D, 1, 1, EPI_ACCUM);

    // MoE
    fira_layernorm<<<MT, 256, 0, stream>>>(x, ln2_w + lD, ln2_b + lD, h, D, 0);
    fira_gate<<<MT, 128, 0, stream>>>(h, gate_w + (long long)l * E * D, gate_b + (long long)l * E,
                                      probs, D, 1.0f / 0.9f);
    (void)hipMemsetAsync(moe, 0, sizeof(float) * (size_t)ACT, stream);
    for (int e = 0; e < E; ++e) {
      const float* w1e = e_w1 + (long long)(l * E + e) * F * D;
      const float* b1e = e_b1 + (long long)(l * E + e) * F;
      const float* w2e = e_w2 + (long long)(l * E + e) * D * F;
      const float* b2e = e_b2 + (long long)(l * E + e) * D;
      gemm(h, D, 0, 0, w1e, D, 0, 0, 0, b1e,
           h1, F, 0, 0, nullptr, 0, MT, F, D, 1, 1, EPI_GELU);
      gemm(h1, F, 0, 0, w2e, F, 0, 0, 0, b2e,
           moe, D, 0, 0, probs + e, E, MT, D, F, 1, 1, EPI_ACCUM);
    }
    fira_layernorm<<<MT, 256, 0, stream>>>(moe, pln_w + lD, pln_b + lD, x, D, 1);
  }

  fira_layernorm<<<MT, 256, 0, stream>>>(x, lnf_w, lnf_b, h, D, 0);
  gemm(h, D, 0, 0, lm_w, D, 0, 0, 0, nullptr,
       (float*)d_out, V, 0, 0, nullptr, 0, MT, V, D, 1, 1, EPI_STORE);
}
